// AbsoluteAttention_79302276153776
// MI455X (gfx1250) — matrix operand order repaired and checked by reading, not compiled, not run on hardware
//
#include <hip/hip_runtime.h>
#include <hip/hip_bf16.h>

typedef __attribute__((ext_vector_type(16))) _Float16 v16h;
typedef __attribute__((ext_vector_type(8)))  float    v8f;

#define SEQ 2048
#define BB  2
#define NH  16
#define DH  64
#define SCALE_F 0.125f
#define MASK_BIG 1e30f

#define ROWSTRIDE (BB*NH*DH)   /* 2048 floats between consecutive seq positions */
#define BSTRIDE   (NH*DH)      /* 1024 floats between batches */

#define SPADI  33              /* padded i-stride in the score plane (floats) */
#define SPLANE (16*SPADI)      /* 528 floats per head plane */

__global__ __launch_bounds__(512)
void xlnet_headsoftmax_attn(const float* __restrict__ q,
                            const float* __restrict__ k,
                            const float* __restrict__ v,
                            const float* __restrict__ mask,
                            float* __restrict__ out) {
    // 16 head planes of raw scores / probabilities for one (16 i) x (32 j) tile
    __shared__ float lds_s[NH * SPLANE];   // 33,792 bytes

    const int tid  = threadIdx.x;
    const int n    = tid >> 5;          // wave index == head index
    const int lane = tid & 31;

    const int b  = blockIdx.x & 1;
    const int i0 = (blockIdx.x >> 1) << 4;

    const int m    = lane & 15;         // A/C row within lane group
    const int kh   = lane >> 4;         // lane half (0/1)
    const int ncol = lane & 15;         // B/C column within lane group
    const int kb16 = kh << 4;           // 0 or 16: K offset for B operand / lane half

    // ---------------- Q A-operands (16 x 32 f16), d-chunks c = 0,1 --------------
    v16h aQ[2];
    {
        const float* qbase = q + (size_t)(i0 + m) * ROWSTRIDE + (size_t)b * BSTRIDE + n * DH;
        #pragma unroll
        for (int c = 0; c < 2; ++c) {
            const float* qp = qbase + c * 32 + kh * 16;
            v16h a;
            #pragma unroll
            for (int h = 0; h < 8; ++h) a[h]     = (_Float16)qp[h];        // K = c*32 + 16*kh + h
            #pragma unroll
            for (int h = 0; h < 8; ++h) a[h + 8] = (_Float16)qp[8 + h];    // K = c*32 + 16*kh + 8 + h
            aQ[c] = a;
        }
    }

    v8f zero8 = {};
    v8f acc[4];                          // O accumulators: d chunks of 16 (N=64 total)
    acc[0] = zero8; acc[1] = zero8; acc[2] = zero8; acc[3] = zero8;

    for (int j0 = 0; j0 < SEQ; j0 += 32) {
        // -------- S = Q * K^T for two 16-wide j sub-tiles (f32 accumulate) ------
        #pragma unroll
        for (int jt = 0; jt < 2; ++jt) {
            const float* kbase = k + (size_t)(j0 + jt * 16 + ncol) * ROWSTRIDE
                                   + (size_t)b * BSTRIDE + n * DH + kb16;
            v8f s = zero8;
            #pragma unroll
            for (int c = 0; c < 2; ++c) {
                const float* kp = kbase + c * 32;
                v16h bK;
                #pragma unroll
                for (int h = 0; h < 16; ++h) bK[h] = (_Float16)kp[h];      // B(k,n): 16 contiguous d
                s = __builtin_amdgcn_wmma_f32_16x16x32_f16(false, aQ[c], false, bK,
                                                           (short)0, s, false, false);
            }
            // C layout: VGPR r -> (M = r + 8*kh, N = ncol)
            float* sp = &lds_s[n * SPLANE + jt * 16 + ncol];
            #pragma unroll
            for (int r = 0; r < 8; ++r) sp[(r + 8 * kh) * SPADI] = s[r];
        }

        __syncthreads();

        // -------- softmax over the HEAD axis n (faithful to reference) ----------
        {
            const int il = tid >> 5;     // 0..15  (i within tile)
            const int jl = tid & 31;     // 0..31  (j within tile)
            const float mval = mask[(size_t)(i0 + il) * (SEQ * BB) + (size_t)(j0 + jl) * BB + b];
            const float bias = -MASK_BIG * mval;   // uniform across n, f32 like reference

            float sv[NH];
            float mx = -3.4e38f;
            float* cell = &lds_s[il * SPADI + jl];
            #pragma unroll
            for (int nn = 0; nn < NH; ++nn) {
                float x = cell[nn * SPLANE] * SCALE_F + bias;
                sv[nn] = x;
                mx = fmaxf(mx, x);
            }
            float sum = 0.f;
            #pragma unroll
            for (int nn = 0; nn < NH; ++nn) {
                float e = __expf(sv[nn] - mx);
                sv[nn] = e;
                sum += e;
            }
            const float inv = 1.0f / sum;
            #pragma unroll
            for (int nn = 0; nn < NH; ++nn) cell[nn * SPLANE] = sv[nn] * inv;
        }

        __syncthreads();

        // -------- P as A-operand (16 x 32 f16, K = j chunk of 32) ---------------
        v16h aP;
        {
            const float* pp = &lds_s[n * SPLANE + m * SPADI + kh * 16];
            #pragma unroll
            for (int h = 0; h < 8; ++h) aP[h]     = (_Float16)pp[h];
            #pragma unroll
            for (int h = 0; h < 8; ++h) aP[h + 8] = (_Float16)pp[8 + h];
        }

        // -------- O += P * V (B operand straight from global/L2) ----------------
        const float* vbase = v + (size_t)(j0 + kb16) * ROWSTRIDE
                               + (size_t)b * BSTRIDE + n * DH + ncol;
        #pragma unroll
        for (int cd = 0; cd < 4; ++cd) {
            const float* vp = vbase + cd * 16;
            v16h bV;
            #pragma unroll
            for (int h = 0; h < 16; ++h) bV[h] = (_Float16)vp[(size_t)h * ROWSTRIDE];
            acc[cd] = __builtin_amdgcn_wmma_f32_16x16x32_f16(false, aP, false, bV,
                                                             (short)0, acc[cd], false, false);
        }

        // prefetch next j-tile's K rows into cache (global_prefetch_b8)
        if (j0 + 32 < SEQ) {
            __builtin_prefetch(k + (size_t)(j0 + 32 + lane) * ROWSTRIDE
                                 + (size_t)b * BSTRIDE + n * DH, 0, 3);
        }

        __syncthreads();   // lds_s is overwritten next iteration
    }

    // ---------------- store O: element (M = r + 8*kh, d = cd*16 + ncol) ---------
    {
        float* obase = out + (size_t)(i0 + 8 * kh) * ROWSTRIDE + (size_t)b * BSTRIDE
                           + n * DH + ncol;
        #pragma unroll
        for (int cd = 0; cd < 4; ++cd) {
            #pragma unroll
            for (int r = 0; r < 8; ++r) {
                obase[(size_t)r * ROWSTRIDE + cd * 16] = acc[cd][r];
            }
        }
    }
}

extern "C" void kernel_launch(void* const* d_in, const int* in_sizes, int n_in,
                              void* d_out, int out_size, void* d_ws, size_t ws_size,
                              hipStream_t stream) {
    (void)in_sizes; (void)n_in; (void)out_size; (void)d_ws; (void)ws_size;
    const float* q    = (const float*)d_in[0];
    const float* k    = (const float*)d_in[1];
    const float* v    = (const float*)d_in[2];
    const float* mask = (const float*)d_in[3];
    float* out        = (float*)d_out;

    dim3 grid((SEQ / 16) * BB);   // 256 blocks: (i-tile, batch)
    dim3 block(512);              // 16 waves = 16 heads per block
    xlnet_headsoftmax_attn<<<grid, block, 0, stream>>>(q, k, v, mask, out);
}
